// PatchDiscriminator_82557861364310
// MI455X (gfx1250) — compile-verified, matrix operand order checked by reading, not run on hardware
//
#include <hip/hip_runtime.h>
#include <hip/hip_bf16.h>

// MI455X (gfx1250) implementation.
// Pipeline (all on `stream`, graph-capture safe, deterministic):
//  1) pack_weights : w1,w2 -> bf16 [tap][cout][cin]; pw -> bf16 [c][t][h]
//  2) layer0       : 1-channel conv + ReLU + norm -> bf16 act (B,L,H), channel-contiguous
//  3) layer_gemm x2: conv-as-GEMM via v_wmma_f32_16x16x32_bf16 (K=cin, 3 shifted taps),
//                    LDS-staged activation slab (async global->LDS when available,
//                    bank-conflict-free padded stride), fused bias+ReLU+scale epilogue
//  4) pool_head    : chunk pooling as second bf16 WMMA GEMM (K = 3*512 over (t,h)),
//                    fused ReLU + fw-dot (shuffle + ds_add_f32) + sigmoid -> d_out
//
// Workspace layout (needs ~261 MB):
//   [0, 128MB)      bufA  bf16 (32,4096,512)
//   [128MB, 256MB)  bufB  bf16 (32,4096,512)
//   [256MB, ...)    packed bf16 weights: wT1, wT2, pwT (786432 elems each)

#define HCH 512
#define SEQ 4096
#define NB  32
#define INV_NORM 0.99999500003749968f  // 1/sqrt(1+1e-5)

// Slab row stride in bf16 elements: 264*2B = 528B = 132 dwords; 132 mod 64 = 4,
// so lanes 0..15 (consecutive positions) hit banks 4i..4i+3 -> conflict-free.
#define SLAB_STRIDE 264

typedef __bf16 bf16;
typedef __attribute__((ext_vector_type(8)))  bf16  bf16x8;
typedef __attribute__((ext_vector_type(16))) bf16  bf16x16;
typedef __attribute__((ext_vector_type(8)))  float v8f;

// Build a v16bf A/B operand from the two contiguous 16B chunks a lane owns:
// elements 0..7  = K (or cin) offset +0, elements 8..15 = offset +16.
static __device__ __forceinline__ bf16x16 ld_pair(const bf16* p) {
  union { bf16x16 v; bf16x8 h[2]; } u;
  u.h[0] = *(const bf16x8*)(p);
  u.h[1] = *(const bf16x8*)(p + 16);
  return u.v;
}

#if defined(__has_builtin)
#if __has_builtin(__builtin_amdgcn_global_load_async_to_lds_b128)
#define HAVE_ASYNC_LDS 1
#endif
#endif

#ifdef HAVE_ASYNC_LDS
// Toolchain signature (from diagnostics): params are pointers to
// `int __attribute__((vector_size(16)))` in AS(1) (global) and AS(3) (LDS).
typedef int i32x4 __attribute__((vector_size(16)));
typedef __attribute__((address_space(1))) i32x4 gi32x4;
typedef __attribute__((address_space(3))) i32x4 li32x4;

static __device__ __forceinline__ void async_copy16(const bf16* g, bf16* l) {
  __builtin_amdgcn_global_load_async_to_lds_b128(
      (gi32x4*)(const_cast<bf16*>(g)), (li32x4*)(l), 0, 0);
}
static __device__ __forceinline__ void wait_async0() {
#if __has_builtin(__builtin_amdgcn_s_wait_asynccnt)
  __builtin_amdgcn_s_wait_asynccnt(0);
#else
  asm volatile("s_wait_asynccnt 0x0" ::: "memory");
#endif
}
#endif

// ---------------------------------------------------------------------------
// 1) Weight packing: f32 -> bf16 with GEMM-friendly permutations.
//    w1/w2: src (cout,cin,k) -> dst ((k*512+cout)*512+cin)
//    pw   : src (c,h,t)      -> dst (c*1536 + t*512 + h)
__global__ void pack_weights(const float* __restrict__ w1, const float* __restrict__ w2,
                             const float* __restrict__ pw,
                             bf16* __restrict__ o1, bf16* __restrict__ o2,
                             bf16* __restrict__ opw) {
  int d = blockIdx.x * 256 + threadIdx.x;           // 0 .. 786431
  int k    = d / (HCH * HCH);
  int rem  = d - k * HCH * HCH;
  int cout = rem >> 9;
  int cin  = rem & 511;
  int s    = (cout * HCH + cin) * 3 + k;
  o1[d] = (bf16)w1[s];
  o2[d] = (bf16)w2[s];
  int c  = d / 1536;
  int r2 = d - c * 1536;
  int t  = r2 >> 9;
  int h  = r2 & 511;
  opw[d] = (bf16)pw[(c * HCH + h) * 3 + t];
}

// ---------------------------------------------------------------------------
// 2) Layer 0: 1-input-channel conv + ReLU + (g*inv)*h + be, write bf16 (B,L,H).
__global__ void layer0_kernel(const float* __restrict__ x, const float* __restrict__ w0,
                              const float* __restrict__ b0, const float* __restrict__ g0,
                              const float* __restrict__ be0, bf16* __restrict__ out) {
  int id = blockIdx.x * 256 + threadIdx.x;          // 32*4096*64 threads
  int cg = id & 63;
  int p  = (id >> 6) & 4095;
  int b  = id >> 18;
  const float* xb = x + (size_t)b * SEQ;
  float xm = (p > 0)        ? xb[p - 1] : 0.f;
  float xc = xb[p];
  float xp = (p < SEQ - 1)  ? xb[p + 1] : 0.f;
  int c0 = cg * 8;
  bf16x8 o;
#pragma unroll
  for (int i = 0; i < 8; ++i) {
    int c = c0 + i;
    float y = w0[c * 3 + 0] * xm + w0[c * 3 + 1] * xc + w0[c * 3 + 2] * xp + b0[c];
    y = fmaxf(y, 0.f) * (g0[c] * INV_NORM) + be0[c];
    o[i] = (bf16)y;
  }
  *(bf16x8*)(out + ((size_t)b * SEQ + p) * HCH + c0) = o;
}

// ---------------------------------------------------------------------------
// 3) Conv layers 1/2 as WMMA GEMM.
//    Tile: 128 cout x 64 positions per workgroup (8 waves, each wave = 16 cout rows,
//    4 accumulators of 16x16). K loop: 2 channel-halves x 3 taps x 8 k-blocks of 32.
//    Activations staged in a 66-position x 256-channel bf16 LDS slab (~34 KB, padded).
__global__ void layer_gemm(const bf16* __restrict__ in, const bf16* __restrict__ wT,
                           const float* __restrict__ bias, const float* __restrict__ gamma,
                           const float* __restrict__ beta, bf16* __restrict__ out) {
  const int ptile = blockIdx.x;                     // 0..63
  const int ctile = blockIdx.y;                     // 0..3
  const int b     = blockIdx.z;                     // 0..31
  const int tid   = threadIdx.x;
  const int wave  = tid >> 5;
  const int lane  = tid & 31;
  const int lhalf = lane >> 4;                      // which half-wave
  const int l16   = lane & 15;
  const int p0        = ptile * 64;
  const int cout_base = ctile * 128 + wave * 16;

  __shared__ bf16 slab[66 * SLAB_STRIDE];           // positions p0-1..p0+64, 256 channels

  v8f acc[4] = {};

  for (int chhalf = 0; chhalf < 2; ++chhalf) {
    const int chbase = chhalf * 256;
    __syncthreads();
    // cooperative slab fill: 66 pos x 32 vec8 of bf16 (16B per thread-chunk)
    for (int v = tid; v < 66 * 32; v += 256) {
      int pos = v >> 5;
      int chv = (v & 31) << 3;
      int gp  = p0 - 1 + pos;
      bf16* dst = slab + pos * SLAB_STRIDE + chv;
      if (gp >= 0 && gp < SEQ) {
        const bf16* src = in + ((size_t)b * SEQ + gp) * HCH + chbase + chv;
#ifdef HAVE_ASYNC_LDS
        async_copy16(src, dst);                     // global -> LDS, no VGPR round trip
#else
        *(bf16x8*)dst = *(const bf16x8*)src;
#endif
      } else {
        bf16x8 z = {};
        *(bf16x8*)dst = z;                          // zero halo (conv padding)
      }
    }
#ifdef HAVE_ASYNC_LDS
    wait_async0();
#endif
    __syncthreads();

    for (int k = 0; k < 3; ++k) {
      const bf16* wbase = wT + ((size_t)k * HCH + (cout_base + l16)) * HCH
                             + chbase + lhalf * 8;
      for (int kb = 0; kb < 8; ++kb) {
        // A: weight rows (M=cout, K=cin block of 32); lane owns K {lh*8..+7, 16+lh*8..+7}
        bf16x16 A = ld_pair(wbase + kb * 32);
        const int lch = kb * 32 + lhalf * 8;
        // B: 4 activation column tiles, loaded up-front so ds_loads clause together
        bf16x16 Bt[4];
#pragma unroll
        for (int ns = 0; ns < 4; ++ns)
          Bt[ns] = ld_pair(slab + (ns * 16 + l16 + k) * SLAB_STRIDE + lch);
#pragma unroll
        for (int ns = 0; ns < 4; ++ns)
          acc[ns] = __builtin_amdgcn_wmma_f32_16x16x32_bf16(
              false, A, false, Bt[ns], (short)0, acc[ns], false, false);
      }
    }
  }

  // Fused epilogue: bias + ReLU + (g*inv) + be, convert bf16, contiguous b128 store.
  // D layout: lane holds col N=l16, rows M = r + 8*lhalf (contiguous in cout).
  const int crow = cout_base + lhalf * 8;
  v8f bv  = *(const v8f*)(bias  + crow);
  v8f gv  = *(const v8f*)(gamma + crow);
  v8f bev = *(const v8f*)(beta  + crow);
#pragma unroll
  for (int ns = 0; ns < 4; ++ns) {
    const int p = p0 + ns * 16 + l16;
    bf16x8 o;
#pragma unroll
    for (int r = 0; r < 8; ++r) {
      float y = acc[ns][r] + bv[r];
      y = fmaxf(y, 0.f) * (gv[r] * INV_NORM) + bev[r];
      o[r] = (bf16)y;
    }
    *(bf16x8*)(out + ((size_t)b * SEQ + p) * HCH + crow) = o;
  }
}

// ---------------------------------------------------------------------------
// 4) Chunk pooling + head. z[n,c] = sum_{t,h} h[b, 3n+t, h] * pwT[c, t, h],
//    out = sigmoid(relu(z) . fw + fb). Tile: 64 chunks per workgroup, 4 passes
//    of 128 c (8 waves x 16 rows). K loop: 3 taps x 16 blocks of 32 channels.
__global__ void pool_head(const bf16* __restrict__ h, const bf16* __restrict__ pwT,
                          const float* __restrict__ pb, const float* __restrict__ fw,
                          const float* __restrict__ fb, const int* __restrict__ seq_lens,
                          float* __restrict__ out) {
  const int tile = blockIdx.x;
  const int b    = blockIdx.y;
  const int tid  = threadIdx.x;
  const int wave = tid >> 5;
  const int lane = tid & 31;
  const int lhalf = lane >> 4;
  const int l16   = lane & 15;

  const int l  = seq_lens[b];
  const int nc = (l - 3) / 3 + 1;                   // chunks in this batch
  const int cb = tile * 64;
  if (cb >= nc) return;                             // uniform across block

  int base = 0;                                     // prefix over earlier batches
  for (int bb = 0; bb < b; ++bb) base += (seq_lens[bb] - 3) / 3 + 1;

  __shared__ float partial[64];
  if (tid < 64) partial[tid] = 0.f;
  __syncthreads();

  for (int cpass = 0; cpass < 4; ++cpass) {
    const int cstrip = cpass * 128 + wave * 16;
    v8f acc[4] = {};
    for (int t = 0; t < 3; ++t) {
      for (int kb = 0; kb < 16; ++kb) {
        const int ch0 = kb * 32 + lhalf * 8;
        bf16x16 A = ld_pair(pwT + (size_t)(cstrip + l16) * 1536 + t * 512 + ch0);
        bf16x16 Bt[4];
#pragma unroll
        for (int ns = 0; ns < 4; ++ns) {
          int chunk = cb + ns * 16 + l16;
          int chk   = (chunk < nc) ? chunk : (nc - 1);  // clamp keeps EXEC all-1s
          int pos   = 3 * chk + t;                      // <= 4094 < SEQ
          Bt[ns] = ld_pair(h + ((size_t)b * SEQ + pos) * HCH + ch0);
        }
#pragma unroll
        for (int ns = 0; ns < 4; ++ns)
          acc[ns] = __builtin_amdgcn_wmma_f32_16x16x32_bf16(
              false, A, false, Bt[ns], (short)0, acc[ns], false, false);
      }
    }
    // reduce: relu(z + pb) * fw, sum over the 16 c rows of this strip
    const int crow = cstrip + lhalf * 8;
    v8f pbv = *(const v8f*)(pb + crow);
    v8f fwv = *(const v8f*)(fw + crow);
#pragma unroll
    for (int ns = 0; ns < 4; ++ns) {
      float s = 0.f;
#pragma unroll
      for (int r = 0; r < 8; ++r) {
        float z = acc[ns][r] + pbv[r];
        s += fmaxf(z, 0.f) * fwv[r];
      }
      s += __shfl_down(s, 16, 32);                  // fold rows 8..15 into 0..7 holder
      if (lhalf == 0) atomicAdd(&partial[ns * 16 + l16], s);
    }
  }
  __syncthreads();

  if (tid < 64) {
    int chunk = cb + tid;
    if (chunk < nc) {
      float logit = partial[tid] + fb[0];
      out[base + chunk] = 1.f / (1.f + __expf(-logit));
    }
  }
}

// ---------------------------------------------------------------------------
extern "C" void kernel_launch(void* const* d_in, const int* in_sizes, int n_in,
                              void* d_out, int out_size, void* d_ws, size_t ws_size,
                              hipStream_t stream) {
  const float* x   = (const float*)d_in[0];
  const float* w0  = (const float*)d_in[1];
  const float* b0  = (const float*)d_in[2];
  const float* g0  = (const float*)d_in[3];
  const float* be0 = (const float*)d_in[4];
  const float* w1  = (const float*)d_in[5];
  const float* b1  = (const float*)d_in[6];
  const float* g1  = (const float*)d_in[7];
  const float* be1 = (const float*)d_in[8];
  const float* w2  = (const float*)d_in[9];
  const float* b2  = (const float*)d_in[10];
  const float* g2  = (const float*)d_in[11];
  const float* be2 = (const float*)d_in[12];
  const float* pw  = (const float*)d_in[13];
  const float* pb  = (const float*)d_in[14];
  const float* fw  = (const float*)d_in[15];
  const float* fb  = (const float*)d_in[16];
  const int* seq_lens = (const int*)d_in[17];
  float* out = (float*)d_out;

  char* ws = (char*)d_ws;
  const size_t ACT_BYTES = (size_t)NB * SEQ * HCH * 2;   // 128 MB
  bf16* bufA = (bf16*)ws;
  bf16* bufB = (bf16*)(ws + ACT_BYTES);
  bf16* wT1  = (bf16*)(ws + 2 * ACT_BYTES);
  bf16* wT2  = wT1 + (size_t)HCH * HCH * 3;
  bf16* pwT  = wT2 + (size_t)HCH * HCH * 3;

  pack_weights<<<(HCH * HCH * 3) / 256, 256, 0, stream>>>(w1, w2, pw, wT1, wT2, pwT);
  layer0_kernel<<<(NB * SEQ * 64) / 256, 256, 0, stream>>>(x, w0, b0, g0, be0, bufA);
  layer_gemm<<<dim3(SEQ / 64, HCH / 128, NB), 256, 0, stream>>>(bufA, wT1, b1, g1, be1, bufB);
  layer_gemm<<<dim3(SEQ / 64, HCH / 128, NB), 256, 0, stream>>>(bufB, wT2, b2, g2, be2, bufA);
  pool_head<<<dim3(22, NB), 256, 0, stream>>>(bufA, pwT, pb, fw, fb, seq_lens, out);
}
